// Gtu_72129680769622
// MI455X (gfx1250) — matrix operand order repaired and checked by reading, not compiled, not run on hardware
//
#include <hip/hip_runtime.h>

#define EPSV 1e-6f
#define GAMMAV 0.99f

typedef __attribute__((ext_vector_type(16))) __bf16 v16bf;
typedef __attribute__((ext_vector_type(8)))  float v8f;
typedef int v4i __attribute__((vector_size(16)));   // matches builtin's V4i param

static __device__ __forceinline__ unsigned short f2bf(float a) {
    unsigned ua = __builtin_bit_cast(unsigned, a);
    ua = ua + 0x7fffu + ((ua >> 16) & 1u);
    return (unsigned short)(ua >> 16);
}
static __device__ __forceinline__ unsigned packbf(float a, float b) {
    return (unsigned)f2bf(a) | ((unsigned)f2bf(b) << 16);
}

union Frag { v16bf v; unsigned u[8]; };

#if __has_builtin(__builtin_amdgcn_global_load_async_to_lds_b128)
#define HAVE_ASYNC_LDS 1
#else
#define HAVE_ASYNC_LDS 0
#endif

static __device__ __forceinline__ void async_copy16(const unsigned short* g, unsigned* l) {
#if HAVE_ASYNC_LDS
    __builtin_amdgcn_global_load_async_to_lds_b128(
        (__attribute__((address_space(1))) v4i*)g,
        (__attribute__((address_space(3))) v4i*)l, 0, 0);
#else
    uint4 d = *(const uint4*)g;
    *(uint4*)l = d;
#endif
}
static __device__ __forceinline__ void wait_async_le(int pending) {
#if HAVE_ASYNC_LDS
    if (pending) __builtin_amdgcn_s_wait_asynccnt(2);
    else         __builtin_amdgcn_s_wait_asynccnt(0);
#else
    (void)pending;
#endif
}

// ---------------------------------------------------------------------------
// fp32 -> bf16 packed conversion (x and weight matrices), float4 -> 2 dwords
// ---------------------------------------------------------------------------
__global__ void cvt_bf16(const float* __restrict__ in, unsigned* __restrict__ out2, int n4) {
    int i = blockIdx.x * 256 + threadIdx.x;
    if (i >= n4) return;
    float4 f = ((const float4*)in)[i];
    out2[2 * i]     = packbf(f.x, f.y);
    out2[2 * i + 1] = packbf(f.z, f.w);
}

// ---------------------------------------------------------------------------
// WMMA bf16 GEMM: Out[M,Nn] = epilogue( A[M,K](bf16) @ Bw[K,Nn](bf16) + bias )
// Block tile 128x64, 8 waves, each wave 32x32 (2x2 WMMA accumulators).
// A tiles stream global->LDS via async copies, 3-stage LDS ring.
// ---------------------------------------------------------------------------
__global__ __launch_bounds__(256) void gemm_bf16(
    const unsigned short* __restrict__ A, const unsigned short* __restrict__ Bw,
    const float* __restrict__ bias, float* __restrict__ Out,
    int M, int Nn, int K, int doSilu)
{
    __shared__ unsigned lA[3][128 * 16];  // [stage][m][kpair] 3x8 KB
    __shared__ unsigned lB[3][64 * 16];   // [stage][n][kpair] 3x4 KB

    const int tid  = threadIdx.x;
    const int lane = tid & 31;
    const int wave = tid >> 5;
    const int wm   = wave & 3;
    const int wn   = wave >> 2;
    const int bM   = blockIdx.x * 128;
    const int bN   = blockIdx.y * 64;
    const int ml   = lane & 15;
    const int half = lane >> 4;
    const int steps = K >> 5;
    const unsigned* Bd = (const unsigned*)Bw;   // dword = 2 consecutive n

    v8f acc[2][2];
#pragma unroll
    for (int i = 0; i < 2; ++i)
#pragma unroll
        for (int j = 0; j < 2; ++j)
            acc[i][j] = (v8f){0.f,0.f,0.f,0.f,0.f,0.f,0.f,0.f};

    auto issueStage = [&](int k0, int s) {
        // A tile 128x32 bf16 (8 KB): 512 x 16B lines, 2 async ops per thread
#pragma unroll
        for (int p = 0; p < 2; ++p) {
            int q = tid + 256 * p;
            int row = q >> 2, seg = q & 3;
            async_copy16(A + (size_t)(bM + row) * K + k0 + seg * 8,
                         &lA[s][row * 16 + seg * 4]);
        }
        // B tile 32x64 bf16 (4 KB), transpose into [n][k] ushorts
        unsigned short* lBs = (unsigned short*)lB[s];
#pragma unroll
        for (int p = 0; p < 4; ++p) {
            int q = tid + 256 * p;          // 1024 dwords
            int krow = q >> 5, nd = q & 31;
            unsigned d = Bd[(size_t)(k0 + krow) * (Nn >> 1) + nd];
            lBs[(nd * 2 + 0) * 32 + krow] = (unsigned short)(d & 0xffffu);
            lBs[(nd * 2 + 1) * 32 + krow] = (unsigned short)(d >> 16);
        }
    };

    issueStage(0, 0);
    for (int it = 0; it < steps; ++it) {
        const int nxt = it + 1;
        if (nxt < steps) issueStage(nxt << 5, nxt % 3);
        wait_async_le(nxt < steps);
        __syncthreads();

        const unsigned* pA = lA[it % 3];
        const unsigned* pB = lB[it % 3];
        Frag a0, a1, b0, b1;
        const int ra0 = (wm * 32 + ml) * 16;
        const int ra1 = (wm * 32 + 16 + ml) * 16;
        const int rb0 = (wn * 32 + ml) * 16;
        const int rb1 = (wn * 32 + 16 + ml) * 16;
#pragma unroll
        for (int j = 0; j < 8; ++j) {
            int ka = half * 4 + j + ((j >= 4) ? 4 : 0);   // lanes0-15 K{0-7,16-23}
            a0.u[j] = pA[ra0 + ka];
            a1.u[j] = pA[ra1 + ka];
            b0.u[j] = pB[rb0 + ka];                       // lanes0-15 K{0-7,16-23}
            b1.u[j] = pB[rb1 + ka];
        }
        acc[0][0] = __builtin_amdgcn_wmma_f32_16x16x32_bf16(false, a0.v, false, b0.v, (short)0, acc[0][0], false, false);
        acc[0][1] = __builtin_amdgcn_wmma_f32_16x16x32_bf16(false, a0.v, false, b1.v, (short)0, acc[0][1], false, false);
        acc[1][0] = __builtin_amdgcn_wmma_f32_16x16x32_bf16(false, a1.v, false, b0.v, (short)0, acc[1][0], false, false);
        acc[1][1] = __builtin_amdgcn_wmma_f32_16x16x32_bf16(false, a1.v, false, b1.v, (short)0, acc[1][1], false, false);
    }

    // C layout: lane L, VGPR r -> (M = r + 8*(L/16), N = L%16)
#pragma unroll
    for (int tm = 0; tm < 2; ++tm)
#pragma unroll
        for (int tn = 0; tn < 2; ++tn) {
            int n = bN + wn * 32 + tn * 16 + ml;
            float bb = bias[n];
#pragma unroll
            for (int r = 0; r < 8; ++r) {
                int m = bM + wm * 32 + tm * 16 + half * 8 + r;
                float v = acc[tm][tn][r] + bb;
                if (doSilu) v = v / (1.0f + __expf(-v));
                Out[(size_t)m * Nn + n] = v;
            }
        }
}

// ---------------------------------------------------------------------------
// RPE position MLP kernels (tiny: ~3 GFLOP total)
// ---------------------------------------------------------------------------
__global__ void rpe_init(const float* __restrict__ Win, const float* __restrict__ bin,
                         float* __restrict__ h)
{
    int idx = blockIdx.x * 256 + threadIdx.x;   // N*R = 2048*512
    int n = idx >> 9, r = idx & 511;
    h[idx] = (float)n * Win[r] + bin[r];
}

__global__ __launch_bounds__(256) void rpe_layer(
    const float* __restrict__ hin, const float* __restrict__ W,
    const float* __restrict__ b, float* __restrict__ hout)
{
    __shared__ float g[512];
    __shared__ float sred[256];
    const int row = blockIdx.x, tid = threadIdx.x;
    const float* hr = hin + row * 512;
    float ss = 0.f;
    for (int r = tid; r < 512; r += 256) { float v = hr[r]; ss += v * v; }
    sred[tid] = ss; __syncthreads();
    for (int s = 128; s > 0; s >>= 1) { if (tid < s) sred[tid] += sred[tid + s]; __syncthreads(); }
    float scale = 1.0f / (sqrtf(sred[0] * (1.0f / 512.0f)) + EPSV);
    for (int r = tid; r < 512; r += 256) { float v = hr[r] * scale; g[r] = v > 0.f ? v : 0.f; }
    __syncthreads();
    for (int rc = tid; rc < 512; rc += 256) {
        float acc = b[rc];
        for (int k = 0; k < 512; ++k) acc += g[k] * W[k * 512 + rc];
        hout[row * 512 + rc] = acc;
    }
}

__global__ __launch_bounds__(256) void rpe_out(
    const float* __restrict__ hin, const float* __restrict__ Wout,
    const float* __restrict__ bout, float* __restrict__ Af)
{
    __shared__ float g[512];
    __shared__ float sred[256];
    const int row = blockIdx.x, tid = threadIdx.x;
    const float* hr = hin + row * 512;
    float ss = 0.f;
    for (int r = tid; r < 512; r += 256) { float v = hr[r]; ss += v * v; }
    sred[tid] = ss; __syncthreads();
    for (int s = 128; s > 0; s >>= 1) { if (tid < s) sred[tid] += sred[tid + s]; __syncthreads(); }
    float scale = 1.0f / (sqrtf(sred[0] * (1.0f / 512.0f)) + EPSV);
    for (int r = tid; r < 512; r += 256) { float v = hr[r] * scale; g[r] = v > 0.f ? v : 0.f; }
    __syncthreads();
    float dec = (row == 0) ? 1.0f : __powf(GAMMAV, (float)row);
    for (int rc = tid; rc < 1536; rc += 256) {
        float acc = bout[rc];
        for (int k = 0; k < 512; ++k) acc += g[k] * Wout[k * 1536 + rc];
        Af[(size_t)row * 1536 + rc] = acc * dec;
    }
}

// ---------------------------------------------------------------------------
// Depthwise causal decayed conv + fused gate, bf16 output:
//   Yg[b,n,c] = bf16( U[b,n,c] * sum_{t=0..n} Af[t,c] * V[b,n-t,c] )
// ---------------------------------------------------------------------------
__global__ __launch_bounds__(256) void conv_depthwise(
    const float* __restrict__ V, const float* __restrict__ Af,
    const float* __restrict__ U, unsigned short* __restrict__ Yg)
{
    constexpr int N = 2048, C = 1536;
    __shared__ float Vt[32 * 64];     // 8 KB
    __shared__ float At[63 * 64];     // ~16 KB filter band
    const int I  = blockIdx.x;
    const int c0 = blockIdx.y * 64;
    const int b  = blockIdx.z;
    const int tid = threadIdx.x;
    const int c  = tid & 63;
    const int i0 = (tid >> 6) * 8;

    float acc[8];
#pragma unroll
    for (int ii = 0; ii < 8; ++ii) acc[ii] = 0.f;

    for (int J = 0; J <= I; ++J) {
        __syncthreads();
        for (int q = tid; q < 32 * 64; q += 256) {
            int j = q >> 6, cc = q & 63;
            Vt[q] = V[(size_t)(b * N + J * 32 + j) * C + c0 + cc];
        }
        int dbase = (I - J) * 32 - 31;
        for (int q = tid; q < 63 * 64; q += 256) {
            int k = q >> 6, cc = q & 63;
            int tau = dbase + k;
            At[q] = (tau >= 0) ? Af[(size_t)tau * C + c0 + cc] : 0.0f;
        }
        __syncthreads();
#pragma unroll 4
        for (int j = 0; j < 32; ++j) {
            float vv = Vt[j * 64 + c];
            int kb = (31 + i0 - j) * 64 + c;
#pragma unroll
            for (int ii = 0; ii < 8; ++ii)
                acc[ii] += At[kb + ii * 64] * vv;
        }
    }
#pragma unroll
    for (int ii = 0; ii < 8; ++ii) {
        size_t idx = (size_t)(b * N + I * 32 + i0 + ii) * C + c0 + c;
        Yg[idx] = f2bf(acc[ii] * U[idx]);
    }
}

// ---------------------------------------------------------------------------
extern "C" void kernel_launch(void* const* d_in, const int* in_sizes, int n_in,
                              void* d_out, int out_size, void* d_ws, size_t ws_size,
                              hipStream_t stream) {
    (void)in_sizes; (void)n_in; (void)out_size; (void)ws_size;
    const float* x    = (const float*)d_in[0];   // (8,2048,512)
    const float* Wu   = (const float*)d_in[1];   // (512,1536)
    const float* bu   = (const float*)d_in[2];
    const float* Wv   = (const float*)d_in[3];
    const float* bv   = (const float*)d_in[4];
    const float* Wo   = (const float*)d_in[5];   // (1536,512)
    const float* bo   = (const float*)d_in[6];
    const float* Win  = (const float*)d_in[7];
    const float* bin  = (const float*)d_in[8];
    const float* rW   = (const float*)d_in[9];   // (3,512,512)
    const float* rb   = (const float*)d_in[10];
    const float* Wout = (const float*)d_in[11];  // (512,1536)
    const float* bout = (const float*)d_in[12];
    float* out = (float*)d_out;                  // (8,2048,512)

    const int M = 8 * 2048;                      // 16384 rows
    char* ws = (char*)d_ws;
    size_t o = 0;
    auto carve = [&](size_t bytes) { char* p = ws + o; o += (bytes + 63) & ~(size_t)63; return p; };
    float*          U   = (float*)carve((size_t)M * 1536 * 4);           // silu(x@Wu+bu)
    float*          Vv  = (float*)carve((size_t)M * 1536 * 4);           // silu(x@Wv+bv)
    float*          Af  = (float*)carve((size_t)2048 * 1536 * 4);        // decayed coeffs
    unsigned short* Yg  = (unsigned short*)carve((size_t)M * 1536 * 2);  // bf16 gated conv out
    unsigned short* xbf = (unsigned short*)carve((size_t)M * 512 * 2);
    unsigned short* Wub = (unsigned short*)carve((size_t)512 * 1536 * 2);
    unsigned short* Wvb = (unsigned short*)carve((size_t)512 * 1536 * 2);
    unsigned short* Wob = (unsigned short*)carve((size_t)1536 * 512 * 2);
    float* hA = (float*)carve((size_t)2048 * 512 * 4);
    float* hB = (float*)carve((size_t)2048 * 512 * 4);

    // one-time bf16 conversions
    cvt_bf16<<<(M * 512 / 4 + 255) / 256, 256, 0, stream>>>(x, (unsigned*)xbf, M * 512 / 4);
    cvt_bf16<<<(512 * 1536 / 4 + 255) / 256, 256, 0, stream>>>(Wu, (unsigned*)Wub, 512 * 1536 / 4);
    cvt_bf16<<<(512 * 1536 / 4 + 255) / 256, 256, 0, stream>>>(Wv, (unsigned*)Wvb, 512 * 1536 / 4);
    cvt_bf16<<<(1536 * 512 / 4 + 255) / 256, 256, 0, stream>>>(Wo, (unsigned*)Wob, 1536 * 512 / 4);

    // RPE MLP
    rpe_init<<<(2048 * 512) / 256, 256, 0, stream>>>(Win, bin, hA);
    rpe_layer<<<2048, 256, 0, stream>>>(hA, rW + 0 * 512 * 512, rb + 0 * 512, hB);
    rpe_layer<<<2048, 256, 0, stream>>>(hB, rW + 1 * 512 * 512, rb + 1 * 512, hA);
    rpe_layer<<<2048, 256, 0, stream>>>(hA, rW + 2 * 512 * 512, rb + 2 * 512, hB);
    rpe_out<<<2048, 256, 0, stream>>>(hB, Wout, bout, Af);

    // u, v projections with SiLU (WMMA bf16, async-pipelined A tiles)
    gemm_bf16<<<dim3(M / 128, 1536 / 64), 256, 0, stream>>>(xbf, Wub, bu, U, M, 1536, 512, 1);
    gemm_bf16<<<dim3(M / 128, 1536 / 64), 256, 0, stream>>>(xbf, Wvb, bv, Vv, M, 1536, 512, 1);

    // depthwise decayed causal conv + fused gating, bf16 output
    conv_depthwise<<<dim3(2048 / 32, 1536 / 64, 8), 256, 0, stream>>>(Vv, Af, U, Yg);

    // output projection: out = Yg @ Wo + bo
    gemm_bf16<<<dim3(M / 128, 512 / 64), 256, 0, stream>>>(Yg, Wob, bo, out, M, 512, 1536, 0);
}
